// MyLSTMCell_48146583388298
// MI455X (gfx1250) — matrix operand order repaired and checked by reading, not compiled, not run on hardware
//
#include <hip/hip_runtime.h>

typedef _Float16 v16h __attribute__((ext_vector_type(16)));
typedef _Float16 v8h  __attribute__((ext_vector_type(8)));
typedef float    v8f  __attribute__((ext_vector_type(8)));

#define B_SZ 4096
#define D_SZ 1024
#define H_SZ 1024
#define KCAT 2048   // D + H concatenated K
#define BK   32
#define NT   (KCAT / BK)   // 64 K-tiles
#define BM   128
#define BN   32
#define SA   40     // padded f16 row stride for A tile (80B = 20 banks)
#define SB   40     // padded f16 row stride for B tile

__device__ __forceinline__ float fsig(float x)  { return 1.0f / (1.0f + __expf(-x)); }
__device__ __forceinline__ float ftanh_(float x){ return 2.0f / (1.0f + __expf(-2.0f * x)) - 1.0f; }

// gfx1250 async global->LDS copy, tracked by ASYNCcnt (no VGPR staging).
__device__ __forceinline__ void async_copy_b128(unsigned lds_byte_addr, const void* gaddr) {
    asm volatile("global_load_async_to_lds_b128 %0, %1, off"
                 :: "v"(lds_byte_addr), "v"(gaddr)
                 : "memory");
}
__device__ __forceinline__ unsigned lds_off(const void* p) {
    // generic LDS pointer: low 32 bits are the LDS byte address
    return (unsigned)(uintptr_t)p;
}

// ---------------------------------------------------------------------------
// Pre-pass 1: pack [X | h_prev] row-major into f16 Acat[4096][2048]
// ---------------------------------------------------------------------------
__global__ __launch_bounds__(256) void conv_act(const float* __restrict__ X,
                                                const float* __restrict__ Hp,
                                                _Float16* __restrict__ Acat) {
    int t  = blockIdx.x * 256 + threadIdx.x;
    int r  = t >> 8;              // 2048/8 = 256 chunks per row
    int kc = (t & 255) << 3;
    const float* src = (kc < D_SZ) ? (X  + (size_t)r * D_SZ + kc)
                                   : (Hp + (size_t)r * H_SZ + (kc - D_SZ));
    float4 lo = ((const float4*)src)[0];
    float4 hi = ((const float4*)src)[1];
    union { v8h v; _Float16 e[8]; } o;
    o.e[0] = (_Float16)lo.x; o.e[1] = (_Float16)lo.y;
    o.e[2] = (_Float16)lo.z; o.e[3] = (_Float16)lo.w;
    o.e[4] = (_Float16)hi.x; o.e[5] = (_Float16)hi.y;
    o.e[6] = (_Float16)hi.z; o.e[7] = (_Float16)hi.w;
    *(v8h*)(Acat + (size_t)r * KCAT + kc) = o.v;
}

// ---------------------------------------------------------------------------
// Pre-pass 2: convert + transpose weights into n-major f16 Wt[4][1024][2048]
// Wt[g][n][k] = (k < D) ? W_g[k][n] : U_g[k-D][n]
// ---------------------------------------------------------------------------
__global__ __launch_bounds__(256) void conv_wt(const float* __restrict__ Wf, const float* __restrict__ Uf,
                                               const float* __restrict__ Wi, const float* __restrict__ Ui,
                                               const float* __restrict__ Wc, const float* __restrict__ Uc,
                                               const float* __restrict__ Wo, const float* __restrict__ Uo,
                                               _Float16* __restrict__ Wt) {
    int t   = blockIdx.x * 256 + threadIdx.x;
    int g   = t >> 18;            // 1024 * 256 chunks per gate
    int rem = t & 262143;
    int n   = rem >> 8;
    int kc  = (rem & 255) << 3;
    const float* Wg; const float* Ug;
    switch (g) {
        case 0:  Wg = Wf; Ug = Uf; break;
        case 1:  Wg = Wi; Ug = Ui; break;
        case 2:  Wg = Wc; Ug = Uc; break;
        default: Wg = Wo; Ug = Uo; break;
    }
    const float* src = (kc < D_SZ) ? Wg : Ug;
    int k0 = (kc < D_SZ) ? kc : (kc - D_SZ);
    union { v8h v; _Float16 e[8]; } o;
#pragma unroll
    for (int j = 0; j < 8; ++j)
        o.e[j] = (_Float16)src[(size_t)(k0 + j) * H_SZ + n];
    *(v8h*)(Wt + ((size_t)g * H_SZ + n) * KCAT + kc) = o.v;
}

// ---------------------------------------------------------------------------
// Fused LSTM cell: per 128x32 output tile, accumulate all 4 gate GEMMs over
// K=2048 with v_wmma_f32_16x16x32_f16.  Global->LDS staging uses gfx1250
// async copies (ASYNCcnt), double-buffered LDS, manual 2-tile unroll so the
// steady-state loop is branch-free with compile-time buffer selection.
// ---------------------------------------------------------------------------
__global__ __launch_bounds__(256) void lstm_wmma(
    const _Float16* __restrict__ A,    // [4096][2048] f16
    const _Float16* __restrict__ Wt,   // [4][1024][2048] f16, n-major
    const float* __restrict__ bfp, const float* __restrict__ bip,
    const float* __restrict__ bcp, const float* __restrict__ bop,
    const float* __restrict__ Cp,
    float* __restrict__ Hn, float* __restrict__ Cn) {

    __shared__ _Float16 lA[2][BM * SA];        // 2 x 10240 B
    __shared__ _Float16 lB[2][4 * BN * SB];    // 2 x 10240 B

    const int tid   = threadIdx.x;
    const int lane  = tid & 31;
    const int wave  = tid >> 5;
    const int waveM = wave >> 1;      // 0..3 -> 32-row slab
    const int waveN = wave & 1;       // 0..1 -> 16-col slab
    const int blockM = blockIdx.y * BM;
    const int blockN = blockIdx.x * BN;

    // --- staging maps: each thread async-copies 2x b128 of A, 2x b128 of B ---
    const int aRow0 = tid >> 2;           // rows 0..63
    const int aRow1 = aRow0 + 64;         // rows 64..127
    const int aK    = (tid & 3) << 3;     // k offset within BK
    const int bg0   = tid >> 7;           // gates 0..1
    const int bg1   = bg0 + 2;            // gates 2..3
    const int bn0   = (tid >> 2) & 31;

    const _Float16* gA0 = A + (size_t)(blockM + aRow0) * KCAT + aK;
    const _Float16* gA1 = A + (size_t)(blockM + aRow1) * KCAT + aK;
    const _Float16* gB0 = Wt + ((size_t)bg0 * H_SZ + blockN + bn0) * KCAT + aK;
    const _Float16* gB1 = Wt + ((size_t)bg1 * H_SZ + blockN + bn0) * KCAT + aK;

    unsigned sa0[2], sa1[2], sb0[2], sb1[2];
#pragma unroll
    for (int b = 0; b < 2; ++b) {
        sa0[b] = lds_off(&lA[b][aRow0 * SA + aK]);
        sa1[b] = lds_off(&lA[b][aRow1 * SA + aK]);
        sb0[b] = lds_off(&lB[b][(bg0 * BN + bn0) * SB + aK]);
        sb1[b] = lds_off(&lB[b][(bg1 * BN + bn0) * SB + aK]);
    }

    // --- fragment addressing (ISA 16-bit A 16x32 / B 32x16 layouts) ---
    const int fr   = lane & 15;
    const int khal = lane >> 4;
    const int fAoff = (waveM * 32 + fr) * SA + khal * 8;    // + m*16*SA
    const int fBoff = (waveN * 16 + fr) * SB + khal * 8;    // + g*BN*SB

    const v8f vzero = {0.f, 0.f, 0.f, 0.f, 0.f, 0.f, 0.f, 0.f};
    v8f acc[4][2];
#pragma unroll
    for (int g = 0; g < 4; ++g)
#pragma unroll
        for (int m = 0; m < 2; ++m) acc[g][m] = vzero;

    auto prefetch_tile = [&](int nb, int ko) {
        async_copy_b128(sa0[nb], gA0 + ko);
        async_copy_b128(sa1[nb], gA1 + ko);
        async_copy_b128(sb0[nb], gB0 + ko);
        async_copy_b128(sb1[nb], gB1 + ko);
    };

    auto compute_tile = [&](int buf) {
        v16h af[2];
#pragma unroll
        for (int m = 0; m < 2; ++m) {
            const _Float16* p = &lA[buf][fAoff + m * 16 * SA];
            union { v16h v; v8h h[2]; } u;
            u.h[0] = *(const v8h*)(p);        // K 0-7   (or 8-15)
            u.h[1] = *(const v8h*)(p + 16);   // K 16-23 (or 24-31)
            af[m] = u.v;
        }
        v16h bf_[4];
#pragma unroll
        for (int g = 0; g < 4; ++g) {
            const _Float16* p = &lB[buf][fBoff + g * (BN * SB)];
            union { v16h v; v8h h[2]; } u;
            u.h[0] = *(const v8h*)(p);        // K 0-7   (or 8-15)
            u.h[1] = *(const v8h*)(p + 16);   // K 16-23 (or 24-31)
            bf_[g] = u.v;
        }
#pragma unroll
        for (int g = 0; g < 4; ++g)
#pragma unroll
            for (int m = 0; m < 2; ++m)
                acc[g][m] = __builtin_amdgcn_wmma_f32_16x16x32_f16(
                    false, af[m], false, bf_[g], (short)0, acc[g][m], false, false);
    };

    // prefetch tile 0 into buffer 0
    prefetch_tile(0, 0);

    // steady state: 31 pairs = tiles 0..61, buffer parity compile-time constant
    int ko = BK;
#pragma unroll 1
    for (int t = 0; t < (NT - 2) / 2; ++t) {
        __syncthreads();                      // buf1 free (readers of 2 tiles ago done)
        prefetch_tile(1, ko); ko += BK;
        asm volatile("s_wait_asynccnt 4");    // tile in buf0 landed (own copies)
        __syncthreads();                      // all waves' copies landed
        compute_tile(0);

        __syncthreads();
        prefetch_tile(0, ko); ko += BK;
        asm volatile("s_wait_asynccnt 4");
        __syncthreads();
        compute_tile(1);
    }
    // tile NT-2 (buf0): prefetch last tile into buf1
    __syncthreads();
    prefetch_tile(1, ko);
    asm volatile("s_wait_asynccnt 4");
    __syncthreads();
    compute_tile(0);
    // tile NT-1 (buf1): drain
    asm volatile("s_wait_asynccnt 0");
    __syncthreads();
    compute_tile(1);

    // --- epilogue: gate nonlinearities + cell update, direct store ---
    const int col = blockN + waveN * 16 + fr;
    const float bias0 = bfp[col];
    const float bias1 = bip[col];
    const float bias2 = bcp[col];
    const float bias3 = bop[col];

#pragma unroll
    for (int m = 0; m < 2; ++m) {
        const int rowBase = blockM + waveM * 32 + m * 16 + khal * 8;
        union { v8f v; float f[8]; } af_, ai_, ac_, ao_;
        af_.v = acc[0][m]; ai_.v = acc[1][m]; ac_.v = acc[2][m]; ao_.v = acc[3][m];
#pragma unroll
        for (int j = 0; j < 8; ++j) {
            const size_t off = (size_t)(rowBase + j) * H_SZ + col;
            float fg = fsig (af_.f[j] + bias0);
            float ig = fsig (ai_.f[j] + bias1);
            float cg = ftanh_(ac_.f[j] + bias2);
            float og = fsig (ao_.f[j] + bias3);
            float cn = fg * Cp[off] + ig * cg;
            Cn[off] = cn;
            Hn[off] = og * ftanh_(cn);
        }
    }
}

// ---------------------------------------------------------------------------
extern "C" void kernel_launch(void* const* d_in, const int* in_sizes, int n_in,
                              void* d_out, int out_size, void* d_ws, size_t ws_size,
                              hipStream_t stream) {
    const float* X   = (const float*)d_in[0];
    const float* Hp  = (const float*)d_in[1];
    const float* Cp  = (const float*)d_in[2];
    const float* Wf  = (const float*)d_in[3];
    const float* Uf  = (const float*)d_in[4];
    const float* bfv = (const float*)d_in[5];
    const float* Wi  = (const float*)d_in[6];
    const float* Ui  = (const float*)d_in[7];
    const float* biv = (const float*)d_in[8];
    const float* Wc  = (const float*)d_in[9];
    const float* Uc  = (const float*)d_in[10];
    const float* bcv = (const float*)d_in[11];
    const float* Wo  = (const float*)d_in[12];
    const float* Uo  = (const float*)d_in[13];
    const float* bov = (const float*)d_in[14];

    _Float16* Acat = (_Float16*)d_ws;                       // 16 MB
    _Float16* Wt   = Acat + (size_t)B_SZ * KCAT;            // 16 MB
    float* Hn = (float*)d_out;
    float* Cn = Hn + (size_t)B_SZ * H_SZ;

    conv_act<<<(B_SZ * (KCAT / 8)) / 256, 256, 0, stream>>>(X, Hp, Acat);
    conv_wt <<<(4 * H_SZ * (KCAT / 8)) / 256, 256, 0, stream>>>(
        Wf, Uf, Wi, Ui, Wc, Uc, Wo, Uo, Wt);

    dim3 grid(H_SZ / BN, B_SZ / BM);   // (32, 32)
    lstm_wmma<<<grid, 256, 0, stream>>>(Acat, Wt, bfv, biv, bcv, bov, Cp, Hn, Cn);
}
